// MemTransformerEncoderLayer_52647709114478
// MI455X (gfx1250) — compile-verified, matrix operand order checked by reading, not run on hardware
//
#include <hip/hip_runtime.h>
#include <hip/hip_bf16.h>

// ---------------- problem constants ----------------
#define T_   256
#define B_   32
#define D_   1024
#define H_   16
#define FF_  4096
#define DH_  64
#define NR_  11            // 2K+1 relative-position bins
#define TB_  (T_ * B_)     // 8192 tokens

// ---------------- vector types ----------------
typedef __attribute__((ext_vector_type(16))) __bf16          bf16x16;
typedef __attribute__((ext_vector_type(8)))  float           f32x8;
typedef __attribute__((ext_vector_type(8)))  unsigned short  u16x8;
typedef __attribute__((ext_vector_type(16))) unsigned short  u16x16;
typedef __attribute__((ext_vector_type(4)))  int             i32x4;
typedef __attribute__((address_space(1)))    i32x4           i32x4_glb;
typedef __attribute__((address_space(3)))    i32x4           i32x4_lds;

__device__ __forceinline__ unsigned short f32_to_bf16_bits(float f) {
    unsigned int u = __builtin_bit_cast(unsigned int, f);
    unsigned int lsb = (u >> 16) & 1u;
    u += 0x7fffu + lsb;                 // round-to-nearest-even
    return (unsigned short)(u >> 16);
}
__device__ __forceinline__ float bf16_bits_to_f32(unsigned short b) {
    unsigned int u = ((unsigned int)b) << 16;
    return __builtin_bit_cast(float, u);
}

// ---------------- CDNA5 async global->LDS copy (ASYNCcnt-tracked) ----------------
__device__ __forceinline__ void async_copy_b128(const unsigned short* g,
                                                const unsigned short* l) {
#if __has_builtin(__builtin_amdgcn_global_load_async_to_lds_b128)
    __builtin_amdgcn_global_load_async_to_lds_b128(
        reinterpret_cast<i32x4_glb*>(reinterpret_cast<uintptr_t>(g)),
        reinterpret_cast<i32x4_lds*>((unsigned)reinterpret_cast<uintptr_t>(l)),
        0, 0);
#else
    unsigned loff = (unsigned)(uintptr_t)l;     // LDS byte address in a VGPR
    asm volatile("global_load_async_to_lds_b128 %0, %1, off"
                 :: "v"(loff), "v"(g) : "memory");
#endif
}

template <int N>
__device__ __forceinline__ void wait_asynccnt() {
#if __has_builtin(__builtin_amdgcn_s_wait_asynccnt)
    __builtin_amdgcn_s_wait_asynccnt((unsigned short)N);
#else
    asm volatile("s_wait_asynccnt %0" :: "i"(N) : "memory");
#endif
}

// ======================================================================
// Generic tiled bf16 GEMM:  C[m,n] = sum_k A[m,k] * Bn[n,k]
//   A  : [batch][M][lda]  bf16 (row-major, K contiguous)
//   Bn : [batch][N][ldb]  bf16 (B transposed: rows are output columns)
//   EPI==0 : C f32 [batch][M][ldc]
//   EPI==1 : C bf16, += bias[n], ReLU  (FFN1)
// Tile 128x128x32, 256 threads = 8 waves (4 along M x 2 along N),
// each wave owns a 32x64 sub-tile = 2x4 WMMA accumulators.
// Double-buffered LDS filled with GLOBAL_LOAD_ASYNC_TO_LDS_B128;
// the next K-slice's async copies overlap the current slice's WMMAs.
// Out-of-range rows are clamped (not zero-filled): those LDS rows only feed
// accumulator slots whose stores are bounds-guarded.
// ======================================================================
#define TM 128
#define TN 128
#define TK 32
#define LSTR 40   // padded LDS row stride (bf16 elems) to spread banks

template <int EPI>
__global__ __launch_bounds__(256) void gemm_bf16_kernel(
    const unsigned short* __restrict__ A,
    const unsigned short* __restrict__ Bn,
    void* __restrict__ C,
    const float* __restrict__ bias,
    int M, int N, int Kdim, int lda, int ldb, int ldc,
    long long strideA, long long strideB, long long strideC)
{
    __shared__ unsigned short As[2][TM * LSTR];
    __shared__ unsigned short Bs[2][TN * LSTR];

    const int batch  = blockIdx.y;
    const int ntiles = (N + TN - 1) / TN;
    const int tileM  = (blockIdx.x / ntiles) * TM;
    const int tileN  = (blockIdx.x % ntiles) * TN;

    A  += (long long)batch * strideA;
    Bn += (long long)batch * strideB;

    const int tid   = threadIdx.x;
    const int wave  = tid >> 5;
    const int lane  = tid & 31;
    const int waveM = (wave & 3) * 32;   // 4 waves along M
    const int waveN = (wave >> 2) * 64;  // 2 waves along N

    // per-thread 16B chunk coordinates inside the 128x32 tile (two chunks each)
    const int rowT0 = (tid * 8) / TK,        colT0 = (tid * 8) % TK;
    const int rowT1 = ((tid + 256) * 8) / TK, colT1 = ((tid + 256) * 8) % TK;
    // clamped global rows
    const int garA0 = min(tileM + rowT0, M - 1);
    const int garA1 = min(tileM + rowT1, M - 1);
    const int garB0 = min(tileN + rowT0, N - 1);
    const int garB1 = min(tileN + rowT1, N - 1);

    f32x8 acc[2][4];
    #pragma unroll
    for (int i = 0; i < 2; ++i)
        #pragma unroll
        for (int j = 0; j < 4; ++j)
            #pragma unroll
            for (int e = 0; e < 8; ++e) acc[i][j][e] = 0.f;

    auto issue_tile = [&](int buf, int k0) {
        async_copy_b128(A + (long long)garA0 * lda + k0 + colT0, &As[buf][rowT0 * LSTR + colT0]);
        async_copy_b128(A + (long long)garA1 * lda + k0 + colT1, &As[buf][rowT1 * LSTR + colT1]);
        async_copy_b128(Bn + (long long)garB0 * ldb + k0 + colT0, &Bs[buf][rowT0 * LSTR + colT0]);
        async_copy_b128(Bn + (long long)garB1 * ldb + k0 + colT1, &Bs[buf][rowT1 * LSTR + colT1]);
    };

    const int nsteps = Kdim / TK;
    issue_tile(0, 0);

    for (int s = 0; s < nsteps; ++s) {
        const int  cur  = s & 1;
        const bool more = (s + 1) < nsteps;
        if (more) issue_tile(cur ^ 1, (s + 1) * TK);
        if (s + 2 < nsteps) {   // speculative L2 prefetch two slices ahead
            __builtin_prefetch(A + (long long)garA0 * lda + (s + 2) * TK + colT0, 0, 1);
            __builtin_prefetch(Bn + (long long)garB0 * ldb + (s + 2) * TK + colT0, 0, 1);
        }
        if (more) wait_asynccnt<4>(); else wait_asynccnt<0>();
        __syncthreads();        // all waves' async fills for `cur` visible

        // ---- build fragments per ISA 16-bit A layout:
        // lane<16 : row = lane,      K = {0..7, 16..23}
        // lane>=16: row = lane-16,   K = {8..15, 24..31}
        const int r     = lane & 15;
        const int khalf = (lane >> 4) * 8;

        bf16x16 afrag[2], bfrag[4];
        #pragma unroll
        for (int i = 0; i < 2; ++i) {
            const unsigned short* p = &As[cur][(waveM + i * 16 + r) * LSTR + khalf];
            u16x8 lo = *(const u16x8*)p;
            u16x8 hi = *(const u16x8*)(p + 16);
            u16x16 bits;
            #pragma unroll
            for (int e = 0; e < 8; ++e) { bits[e] = lo[e]; bits[8 + e] = hi[e]; }
            afrag[i] = __builtin_bit_cast(bf16x16, bits);
        }
        #pragma unroll
        for (int j = 0; j < 4; ++j) {
            const unsigned short* p = &Bs[cur][(waveN + j * 16 + r) * LSTR + khalf];
            u16x8 lo = *(const u16x8*)p;
            u16x8 hi = *(const u16x8*)(p + 16);
            u16x16 bits;
            #pragma unroll
            for (int e = 0; e < 8; ++e) { bits[e] = lo[e]; bits[8 + e] = hi[e]; }
            bfrag[j] = __builtin_bit_cast(bf16x16, bits);
        }

        #pragma unroll
        for (int i = 0; i < 2; ++i)
            #pragma unroll
            for (int j = 0; j < 4; ++j)
                acc[i][j] = __builtin_amdgcn_wmma_f32_16x16x32_bf16(
                    false, afrag[i], false, bfrag[j],
                    (short)0, acc[i][j], false, false);

        __syncthreads();        // done reading `cur` before it is refilled
    }

    // ---- epilogue: C/D layout lane<16 -> (M=reg, N=lane); lane>=16 -> (M=reg+8, N=lane-16)
    const int nlane = lane & 15;
    const int mofs  = (lane >> 4) * 8;
    #pragma unroll
    for (int i = 0; i < 2; ++i) {
        #pragma unroll
        for (int j = 0; j < 4; ++j) {
            int gn = tileN + waveN + j * 16 + nlane;
            #pragma unroll
            for (int reg = 0; reg < 8; ++reg) {
                int gm = tileM + waveM + i * 16 + mofs + reg;
                if (gm < M && gn < N) {
                    float v = acc[i][j][reg];
                    if (EPI == 1) {
                        v += bias[gn];
                        v = v > 0.f ? v : 0.f;
                        ((unsigned short*)C)[(long long)batch * strideC +
                                             (long long)gm * ldc + gn] = f32_to_bf16_bits(v);
                    } else {
                        ((float*)C)[(long long)batch * strideC +
                                    (long long)gm * ldc + gn] = v;
                    }
                }
            }
        }
    }
}

// ======================================================================
// Elementwise / reduction kernels
// ======================================================================
__global__ void f32_to_bf16_kernel(const float* __restrict__ in,
                                   unsigned short* __restrict__ out, long long n) {
    long long i = (long long)blockIdx.x * blockDim.x + threadIdx.x;
    if (i < n) out[i] = f32_to_bf16_bits(in[i]);
}

// qf/kf/vf : f32 [TB, D] token-major (t*B+b). Produce:
//   q_bf,k_bf : bf16 [B,H,T,DH]   (q scaled by 1/sqrt(DH), biases added)
//   vT_bf     : bf16 [B,H,DH,T]   (pre-transposed so GEMM Bn layout works)
__global__ void qkv_repack_kernel(const float* __restrict__ qf, const float* __restrict__ kf,
                                  const float* __restrict__ vf,
                                  const float* __restrict__ bq, const float* __restrict__ bk,
                                  const float* __restrict__ bv,
                                  unsigned short* __restrict__ q_bf,
                                  unsigned short* __restrict__ k_bf,
                                  unsigned short* __restrict__ vT_bf) {
    long long i = (long long)blockIdx.x * blockDim.x + threadIdx.x;
    if (i >= (long long)TB_ * D_) return;
    int d = (int)(i % D_);
    long long tok = i / D_;
    int t = (int)(tok / B_), b = (int)(tok % B_);
    int h = d / DH_, dh = d % DH_;
    const float scale = 0.125f;   // 1/sqrt(64)
    long long bh = (long long)b * H_ + h;
    q_bf[(bh * T_ + t) * DH_ + dh]  = f32_to_bf16_bits((qf[i] + bq[d]) * scale);
    k_bf[(bh * T_ + t) * DH_ + dh]  = f32_to_bf16_bits(kf[i] + bk[d]);
    vT_bf[(bh * DH_ + dh) * T_ + t] = f32_to_bf16_bits(vf[i] + bv[d]);
}

// qp[row, r] = sum_d q_bf[row, d] * pk_table[r, h*DH+d]   (row = ((b*H+h)*T+q))
__global__ void qp_kernel(const unsigned short* __restrict__ q_bf,
                          const float* __restrict__ pk, float* __restrict__ qp) {
    long long i = (long long)blockIdx.x * blockDim.x + threadIdx.x;
    const long long total = (long long)B_ * H_ * T_ * NR_;
    if (i >= total) return;
    int r = (int)(i % NR_);
    long long row = i / NR_;
    int h = (int)((row / T_) % H_);
    const unsigned short* qr = q_bf + row * DH_;
    const float* pr = pk + (long long)r * D_ + h * DH_;
    float s = 0.f;
    #pragma unroll 8
    for (int d = 0; d < DH_; ++d) s += bf16_bits_to_f32(qr[d]) * pr[d];
    qp[i] = s;
}

// One block (256 thr) per score row: add gathered qp term, softmax, emit bf16
// attn and 11 per-band sums for the position-value epilogue.
__global__ __launch_bounds__(256) void softmax_kernel(
    const float* __restrict__ scores, const float* __restrict__ qp,
    unsigned short* __restrict__ attn, float* __restrict__ asum) {
    const int row = blockIdx.x;             // ((b*H+h)*T + q)
    const int q   = row % T_;
    const int k   = threadIdx.x;
    __shared__ float red[256];
    __shared__ float bins[NR_];

    int rel = k - q; rel = rel < -5 ? -5 : (rel > 5 ? 5 : rel);
    const int r = rel + 5;
    float s = scores[(long long)row * T_ + k] + qp[(long long)row * NR_ + r];

    red[k] = s; __syncthreads();
    for (int off = 128; off > 0; off >>= 1) {
        if (k < off) red[k] = fmaxf(red[k], red[k + off]);
        __syncthreads();
    }
    const float mx = red[0]; __syncthreads();

    float e = __expf(s - mx);
    red[k] = e; __syncthreads();
    for (int off = 128; off > 0; off >>= 1) {
        if (k < off) red[k] += red[k + off];
        __syncthreads();
    }
    const float a = e / red[0];
    attn[(long long)row * T_ + k] = f32_to_bf16_bits(a);

    if (k < NR_) bins[k] = 0.f;
    __syncthreads();
    atomicAdd(&bins[r], a);
    __syncthreads();
    if (k < NR_) asum[(long long)row * NR_ + k] = bins[k];
}

// ctx_bf[(t*B+b), d] = bf16( ctxf[b,h,t,dh] + sum_r asum[row,r]*pv[r,d] )
__global__ void ctx_posv_kernel(const float* __restrict__ ctxf,
                                const float* __restrict__ asum,
                                const float* __restrict__ pv,
                                unsigned short* __restrict__ ctx_bf) {
    long long i = (long long)blockIdx.x * blockDim.x + threadIdx.x;
    if (i >= (long long)TB_ * D_) return;
    int dh = (int)(i % DH_);
    long long bht = i / DH_;
    int t = (int)(bht % T_);
    long long bh = bht / T_;
    int h = (int)(bh % H_), b = (int)(bh / H_);
    int d = h * DH_ + dh;
    float v = ctxf[i];
    const float* as = asum + bht * NR_;
    #pragma unroll
    for (int r = 0; r < NR_; ++r) v += as[r] * pv[(long long)r * D_ + d];
    ctx_bf[((long long)t * B_ + b) * D_ + d] = f32_to_bf16_bits(v);
}

// out = LayerNorm(a + c + bias) * g + beta ; writes f32 and/or bf16
__global__ __launch_bounds__(256) void add_ln_kernel(
    const float* __restrict__ a, const float* __restrict__ c,
    const float* __restrict__ bias, const float* __restrict__ g,
    const float* __restrict__ beta,
    float* __restrict__ out_f32, unsigned short* __restrict__ out_bf16) {
    const int row = blockIdx.x;
    const int tid = threadIdx.x;
    __shared__ float red[256];

    float vals[4]; float s = 0.f;
    #pragma unroll
    for (int i = 0; i < 4; ++i) {
        int d = tid + i * 256;
        float v = a[(long long)row * D_ + d] + c[(long long)row * D_ + d];
        if (bias) v += bias[d];
        vals[i] = v; s += v;
    }
    red[tid] = s; __syncthreads();
    for (int off = 128; off > 0; off >>= 1) {
        if (tid < off) red[tid] += red[tid + off];
        __syncthreads();
    }
    const float mu = red[0] * (1.0f / D_); __syncthreads();

    float vs = 0.f;
    #pragma unroll
    for (int i = 0; i < 4; ++i) { float dd = vals[i] - mu; vs += dd * dd; }
    red[tid] = vs; __syncthreads();
    for (int off = 128; off > 0; off >>= 1) {
        if (tid < off) red[tid] += red[tid + off];
        __syncthreads();
    }
    const float rstd = rsqrtf(red[0] * (1.0f / D_) + 1e-5f);

    #pragma unroll
    for (int i = 0; i < 4; ++i) {
        int d = tid + i * 256;
        float y = (vals[i] - mu) * rstd * g[d] + beta[d];
        if (out_f32)  out_f32[(long long)row * D_ + d] = y;
        if (out_bf16) out_bf16[(long long)row * D_ + d] = f32_to_bf16_bits(y);
    }
}

// ======================================================================
// Host-side orchestration
// ======================================================================
static inline void* carve(char*& p, size_t bytes) {
    void* r = (void*)p;
    p += (bytes + 255) & ~(size_t)255;
    return r;
}

extern "C" void kernel_launch(void* const* d_in, const int* in_sizes, int n_in,
                              void* d_out, int out_size, void* d_ws, size_t ws_size,
                              hipStream_t stream) {
    const float* src  = (const float*)d_in[0];
    const float* wq   = (const float*)d_in[1];
    const float* wk   = (const float*)d_in[2];
    const float* wv   = (const float*)d_in[3];
    const float* wo   = (const float*)d_in[4];
    const float* bq   = (const float*)d_in[5];
    const float* bk   = (const float*)d_in[6];
    const float* bv   = (const float*)d_in[7];
    const float* bo   = (const float*)d_in[8];
    const float* pk   = (const float*)d_in[9];
    const float* pv   = (const float*)d_in[10];
    const float* w1   = (const float*)d_in[11];
    const float* b1   = (const float*)d_in[12];
    const float* w2   = (const float*)d_in[13];
    const float* b2   = (const float*)d_in[14];
    const float* ln1g = (const float*)d_in[15];
    const float* ln1b = (const float*)d_in[16];
    const float* ln2g = (const float*)d_in[17];
    const float* ln2b = (const float*)d_in[18];
    // d_in[19] == rpr table; index is computed analytically on device.

    char* p = (char*)d_ws;
    const long long nTok = (long long)TB_ * D_;     // 8.4M
    const long long nBH  = (long long)B_ * H_;      // 512
    const long long nRow = nBH * T_;                // 131072

    unsigned short* src_bf = (unsigned short*)carve(p, nTok * 2);
    unsigned short* wq_bf  = (unsigned short*)carve(p, (size_t)D_ * D_ * 2);
    unsigned short* wk_bf  = (unsigned short*)carve(p, (size_t)D_ * D_ * 2);
    unsigned short* wv_bf  = (unsigned short*)carve(p, (size_t)D_ * D_ * 2);
    unsigned short* wo_bf  = (unsigned short*)carve(p, (size_t)D_ * D_ * 2);
    unsigned short* w1_bf  = (unsigned short*)carve(p, (size_t)FF_ * D_ * 2);
    unsigned short* w2_bf  = (unsigned short*)carve(p, (size_t)D_ * FF_ * 2);
    float* qf   = (float*)carve(p, nTok * 4);
    float* kf   = (float*)carve(p, nTok * 4);
    float* vf   = (float*)carve(p, nTok * 4);
    unsigned short* q_bf  = (unsigned short*)carve(p, nTok * 2);
    unsigned short* k_bf  = (unsigned short*)carve(p, nTok * 2);
    unsigned short* vT_bf = (unsigned short*)carve(p, nTok * 2);
    float* qp    = (float*)carve(p, nRow * NR_ * 4);
    float* asum  = (float*)carve(p, nRow * NR_ * 4);
    float* scores = (float*)carve(p, nRow * T_ * 4);          // 134 MB
    unsigned short* attn_bf = (unsigned short*)carve(p, nRow * T_ * 2);
    float* ctxf  = (float*)carve(p, nTok * 4);
    unsigned short* ctx_bf = (unsigned short*)carve(p, nTok * 2);
    float* proj  = (float*)carve(p, nTok * 4);
    float* x_f32 = (float*)carve(p, nTok * 4);
    unsigned short* x_bf = (unsigned short*)carve(p, nTok * 2);
    unsigned short* h_bf = (unsigned short*)carve(p, (size_t)TB_ * FF_ * 2);
    float* y_f32 = (float*)carve(p, nTok * 4);
    (void)ws_size; (void)n_in; (void)in_sizes; (void)out_size;

    auto cdiv = [](long long a, long long b) { return (int)((a + b - 1) / b); };

    // ---- 1. bf16 conversions ----
    f32_to_bf16_kernel<<<cdiv(nTok, 256), 256, 0, stream>>>(src, src_bf, nTok);
    f32_to_bf16_kernel<<<cdiv((long long)D_ * D_, 256), 256, 0, stream>>>(wq, wq_bf, (long long)D_ * D_);
    f32_to_bf16_kernel<<<cdiv((long long)D_ * D_, 256), 256, 0, stream>>>(wk, wk_bf, (long long)D_ * D_);
    f32_to_bf16_kernel<<<cdiv((long long)D_ * D_, 256), 256, 0, stream>>>(wv, wv_bf, (long long)D_ * D_);
    f32_to_bf16_kernel<<<cdiv((long long)D_ * D_, 256), 256, 0, stream>>>(wo, wo_bf, (long long)D_ * D_);
    f32_to_bf16_kernel<<<cdiv((long long)FF_ * D_, 256), 256, 0, stream>>>(w1, w1_bf, (long long)FF_ * D_);
    f32_to_bf16_kernel<<<cdiv((long long)D_ * FF_, 256), 256, 0, stream>>>(w2, w2_bf, (long long)D_ * FF_);

    // ---- 2. QKV projections: [TB,D] @ W^T (W is [N][K] already) ----
    {
        dim3 grid(cdiv(TB_, TM) * cdiv(D_, TN), 1);
        gemm_bf16_kernel<0><<<grid, 256, 0, stream>>>(src_bf, wq_bf, qf, nullptr,
            TB_, D_, D_, D_, D_, D_, 0, 0, 0);
        gemm_bf16_kernel<0><<<grid, 256, 0, stream>>>(src_bf, wk_bf, kf, nullptr,
            TB_, D_, D_, D_, D_, D_, 0, 0, 0);
        gemm_bf16_kernel<0><<<grid, 256, 0, stream>>>(src_bf, wv_bf, vf, nullptr,
            TB_, D_, D_, D_, D_, D_, 0, 0, 0);
    }

    // ---- 3. bias + scale + head repack (+ V transpose) ----
    qkv_repack_kernel<<<cdiv(nTok, 256), 256, 0, stream>>>(qf, kf, vf, bq, bk, bv,
                                                           q_bf, k_bf, vT_bf);

    // ---- 4. relative-position query projections (11 bins) ----
    qp_kernel<<<cdiv(nRow * NR_, 256), 256, 0, stream>>>(q_bf, pk, qp);

    // ---- 5. batched attention scores: q @ k^T per (b,h) ----
    {
        dim3 grid(cdiv(T_, TM) * cdiv(T_, TN), (unsigned)nBH);
        gemm_bf16_kernel<0><<<grid, 256, 0, stream>>>(q_bf, k_bf, scores, nullptr,
            T_, T_, DH_, DH_, DH_, T_,
            (long long)T_ * DH_, (long long)T_ * DH_, (long long)T_ * T_);
    }

    // ---- 6. softmax (+qp gather, band sums) ----
    softmax_kernel<<<(unsigned)nRow, 256, 0, stream>>>(scores, qp, attn_bf, asum);

    // ---- 7. batched attn @ V ----
    {
        dim3 grid(cdiv(T_, TM) * cdiv(DH_, TN), (unsigned)nBH);
        gemm_bf16_kernel<0><<<grid, 256, 0, stream>>>(attn_bf, vT_bf, ctxf, nullptr,
            T_, DH_, T_, T_, T_, DH_,
            (long long)T_ * T_, (long long)DH_ * T_, (long long)T_ * DH_);
    }

    // ---- 8. add position-value term, pack ctx to token-major bf16 ----
    ctx_posv_kernel<<<cdiv(nTok, 256), 256, 0, stream>>>(ctxf, asum, pv, ctx_bf);

    // ---- 9. output projection ----
    {
        dim3 grid(cdiv(TB_, TM) * cdiv(D_, TN), 1);
        gemm_bf16_kernel<0><<<grid, 256, 0, stream>>>(ctx_bf, wo_bf, proj, nullptr,
            TB_, D_, D_, D_, D_, D_, 0, 0, 0);
    }

    // ---- 10. LN1(src + proj + bo) ----
    add_ln_kernel<<<TB_, 256, 0, stream>>>(src, proj, bo, ln1g, ln1b, x_f32, x_bf);

    // ---- 11. FFN1 with fused bias+ReLU -> bf16 ----
    {
        dim3 grid(cdiv(TB_, TM) * cdiv(FF_, TN), 1);
        gemm_bf16_kernel<1><<<grid, 256, 0, stream>>>(x_bf, w1_bf, h_bf, b1,
            TB_, FF_, D_, D_, D_, FF_, 0, 0, 0);
    }

    // ---- 12. FFN2 ----
    {
        dim3 grid(cdiv(TB_, TM) * cdiv(D_, TN), 1);
        gemm_bf16_kernel<0><<<grid, 256, 0, stream>>>(h_bf, w2_bf, y_f32, nullptr,
            TB_, D_, FF_, FF_, FF_, D_, 0, 0, 0);
    }

    // ---- 13. LN2(x + ffn + b2) -> d_out (f32, [T,B,D] == token-major) ----
    add_ln_kernel<<<TB_, 256, 0, stream>>>(x_f32, y_f32, b2, ln2g, ln2b,
                                           (float*)d_out, nullptr);
}
